// CSNN_58274116272651
// MI455X (gfx1250) — matrix operand order repaired and checked by reading, not compiled, not run on hardware
//
#include <hip/hip_runtime.h>

typedef __attribute__((ext_vector_type(16))) _Float16 v16h;
typedef __attribute__((ext_vector_type(8)))  float    v8f;
typedef __attribute__((ext_vector_type(4)))  unsigned int u32x4;

#define CH  128
#define BB  4
#define TT  16

#if __has_builtin(__builtin_amdgcn_global_load_async_to_lds_b128) && \
    __has_builtin(__builtin_amdgcn_s_wait_asynccnt)
#define CSNN_ASYNC_LDS 1
typedef int v4i __attribute__((vector_size(4 * sizeof(int))));
typedef __attribute__((address_space(1))) v4i gv4i_t;   // global int4 vector
typedef __attribute__((address_space(3))) v4i lv4i_t;   // LDS int4 vector
#endif

// ---------------------------------------------------------------------------
// Pack network input [T,B,2,128,128] f32 -> X0 [(t*B+b)][h][w][32] f16 (zero pad c>=2)
__global__ __launch_bounds__(256) void pack_input(const float* __restrict__ x,
                                                  _Float16* __restrict__ X0) {
  long tid = (long)blockIdx.x * 256 + threadIdx.x;   // 64*128*128*32 total
  int c = tid & 31;
  long p = tid >> 5;
  int w = p & 127; p >>= 7;
  int h = p & 127; p >>= 7;
  long n = p;                                        // t*B+b
  float val = 0.f;
  if (c < 2) val = x[((n * 2 + c) * 128 + h) * 128 + w];
  X0[tid] = (_Float16)val;
}

// Repack conv weights [128][CIN][3][3] f32 -> Wp [128][9][CINP] f16 (zero pad)
__global__ __launch_bounds__(256) void pack_w(const float* __restrict__ w,
                                              _Float16* __restrict__ Wp,
                                              int CIN, int CINP) {
  long tid = (long)blockIdx.x * 256 + threadIdx.x;   // 128*9*CINP total
  int c = tid % CINP;
  long p = tid / CINP;
  int tap = p % 9;
  int co  = p / 9;
  float val = 0.f;
  if (c < CIN) val = w[((long)co * CIN + c) * 9 + tap];
  Wp[tid] = (_Float16)val;
}

// BN constants: bnc[c] = g/sqrt(v+eps), bnc[128+c] = b - m*inv
__global__ void bn_prep(const float* __restrict__ g, const float* __restrict__ b,
                        const float* __restrict__ m, const float* __restrict__ v,
                        float* __restrict__ bnc) {
  int c = threadIdx.x;
  if (c >= CH) return;
  float inv = g[c] * rsqrtf(v[c] + 1e-5f);
  bnc[c] = inv;
  bnc[CH + c] = b[c] - m[c] * inv;
}

// ---------------------------------------------------------------------------
// Implicit-GEMM conv3x3 SAME + fused BN via WMMA f32<=f16.
// X: [N][H][W][CINP] f16 ; Wp: [128][9][CINP] f16 ; U out: [N][H][W][128] f32.
// WG = 8 waves. Each wave: 16(co) x 16*NT(pixel) strip, so every weight (A)
// fragment loaded from global is reused by NT WMMAs. WG covers all 128 co over
// a shared LDS halo patch of 3 x (16*NT+2) x CINP halves. Halo patch is staged
// with async global->LDS b128 copies (ASYNCcnt) when available.
template <int CINP, int HH, int NT>
__global__ __launch_bounds__(256) void conv_wmma(const _Float16* __restrict__ X,
                                                 const _Float16* __restrict__ Wp,
                                                 const float* __restrict__ bnc,
                                                 float* __restrict__ U) {
  constexpr int WW   = HH;
  constexpr int COLS = 16 * NT + 2;
  __shared__ __align__(16) _Float16 Xs[3 * COLS * CINP];

  int tile = blockIdx.x;                       // strip of 16*NT consecutive w
  constexpr int stripsPerRow = WW / (16 * NT);
  int w0 = (tile % stripsPerRow) * (16 * NT);
  int h  = (tile / stripsPerRow) % HH;
  int n  = tile / (stripsPerRow * HH);
  int tid = threadIdx.x;

  // Cooperative LDS fill of the halo patch (8 halves = 16B per chunk).
  constexpr int C8 = CINP / 8;
  for (int i = tid; i < 3 * COLS * C8; i += 256) {
    int c8 = (i % C8) * 8;
    int dx = (i / C8) % COLS;
    int dy = (i / C8) / COLS;
    int hh = h + dy - 1, ww = w0 + dx - 1;
    _Float16* lp = &Xs[((dy * COLS) + dx) * CINP + c8];
    bool inb = (hh >= 0 && hh < HH && ww >= 0 && ww < WW);
#ifdef CSNN_ASYNC_LDS
    if (inb) {
      const _Float16* gp = X + ((((long)n * HH + hh) * WW + ww) * CINP + c8);
      __builtin_amdgcn_global_load_async_to_lds_b128(
          (gv4i_t*)(unsigned long long)gp,
          (lv4i_t*)(unsigned int)(unsigned long long)lp, 0, 0);
    } else {
      *(u32x4*)lp = (u32x4){0u, 0u, 0u, 0u};
    }
#else
    u32x4 val = {0u, 0u, 0u, 0u};
    if (inb)
      val = *(const u32x4*)(X + ((((long)n * HH + hh) * WW + ww) * CINP + c8));
    *(u32x4*)lp = val;
#endif
  }
#ifdef CSNN_ASYNC_LDS
  __builtin_amdgcn_s_wait_asynccnt(0);
#endif
  __syncthreads();

  int wave = tid >> 5;
  int lane = tid & 31;
  int cobase = wave * 16;
  int m  = lane & 15;          // A row (co) and B column (pixel within tile)
  int hi = lane >> 4;          // lane-half selector

  v8f acc[NT];
#pragma unroll
  for (int nt = 0; nt < NT; ++nt) acc[nt] = (v8f){};

  const _Float16* wrow = Wp + (long)(cobase + m) * 9 * CINP;

#pragma unroll
  for (int tap = 0; tap < 9; ++tap) {
    const int r = tap / 3, s = tap % 3;
    if (tap < 8) __builtin_prefetch(wrow + (tap + 1) * CINP, 0, 0);
    for (int kc = 0; kc < CINP; kc += 32) {
      // A 16x32 f16 layout: lane holds K = hi*8..+7 and hi*8+16..+23
      union { v16h h; u32x4 q[2]; } A;
      const _Float16* ap = wrow + tap * CINP + kc + hi * 8;
      A.q[0] = *(const u32x4*)(ap);
      A.q[1] = *(const u32x4*)(ap + 16);
#pragma unroll
      for (int nt = 0; nt < NT; ++nt) {
        // B 32x16 f16 layout: lane holds K = hi*8..+7 and hi*8+16..+23
        union { v16h h; u32x4 q[2]; } Bf;
        const _Float16* bp =
            &Xs[((r * COLS) + (nt * 16 + m + s)) * CINP + kc + hi * 8];
        Bf.q[0] = *(const u32x4*)(bp);
        Bf.q[1] = *(const u32x4*)(bp + 16);
        acc[nt] = __builtin_amdgcn_wmma_f32_16x16x32_f16(
            false, A.h, false, Bf.h, (short)0, acc[nt], false, false);
      }
    }
  }

  // BN + store f32 NHWC: VGPR j -> co = cobase + hi*8 + j
  int co0 = cobase + hi * 8;
  const float* inv  = bnc;
  const float* beta = bnc + CH;
  float iv[8], bt[8];
#pragma unroll
  for (int j = 0; j < 8; ++j) { iv[j] = inv[co0 + j]; bt[j] = beta[co0 + j]; }
#pragma unroll
  for (int nt = 0; nt < NT; ++nt) {
    int w = w0 + nt * 16 + m;
    float* up = U + ((((long)n * HH + h) * WW + w) * CH) + co0;
#pragma unroll
    for (int j = 0; j < 8; ++j)
      up[j] = acc[nt][j] * iv[j] + bt[j];
  }
}

// ---------------------------------------------------------------------------
// Fused LIF (tau=2, vth=1, hard reset) + maxpool2x2 (OR of binary spikes).
// U: [n][HH][HH][128] f32 ; Xn: [n][HH/2][HH/2][128] f16 spikes.
template <int HH>
__global__ __launch_bounds__(256) void lif_pool(const float* __restrict__ U,
                                                _Float16* __restrict__ Xn) {
  constexpr int H2 = HH / 2;
  long tid = (long)blockIdx.x * 256 + threadIdx.x;   // (b,ph,pw,c)
  int c = tid & (CH - 1);
  long p = tid >> 7;
  int pw = p % H2; p /= H2;
  int ph = p % H2; p /= H2;
  int b  = (int)p;
  float v0 = 0.f, v1 = 0.f, v2 = 0.f, v3 = 0.f;
  for (int t = 0; t < TT; ++t) {
    int n = t * BB + b;
    const float* u = U + ((((long)n * HH + 2 * ph) * HH + 2 * pw) * CH) + c;
    float x0 = u[0], x1 = u[CH], x2 = u[(long)HH * CH], x3 = u[(long)HH * CH + CH];
    v0 = 0.5f * (v0 + x0); v1 = 0.5f * (v1 + x1);
    v2 = 0.5f * (v2 + x2); v3 = 0.5f * (v3 + x3);
    float s0 = v0 >= 1.f ? 1.f : 0.f, s1 = v1 >= 1.f ? 1.f : 0.f;
    float s2 = v2 >= 1.f ? 1.f : 0.f, s3 = v3 >= 1.f ? 1.f : 0.f;
    float pooled = fmaxf(fmaxf(s0, s1), fmaxf(s2, s3));
    v0 *= (1.f - s0); v1 *= (1.f - s1); v2 *= (1.f - s2); v3 *= (1.f - s3);
    Xn[((((long)n * H2 + ph) * H2 + pw) * CH) + c] = (_Float16)pooled;
  }
}

// ---------------------------------------------------------------------------
// FC: Z[n][k] = sum over (c,h,w) spikes[n][h][w][c] * wfc[k][c*64+h*8+w] + bfc[k]
__global__ __launch_bounds__(256) void fc_kernel(const _Float16* __restrict__ X4,
                                                 const float* __restrict__ wfc,
                                                 const float* __restrict__ bfc,
                                                 float* __restrict__ Z) {
  __shared__ float red[256];
  int out = blockIdx.x;            // 64*10
  int k = out % 10, n = out / 10;
  const _Float16* xb = X4 + (long)n * 8192;
  const float* wk = wfc + (long)k * 8192;
  float s = 0.f;
  for (int i = threadIdx.x; i < 8192; i += 256) {
    int c = i & 127, hw = i >> 7;                 // storage (h,w,c); weight c*64+hw
    s += (float)xb[i] * wk[c * 64 + hw];
  }
  red[threadIdx.x] = s;
  __syncthreads();
  for (int o = 128; o > 0; o >>= 1) {
    if (threadIdx.x < o) red[threadIdx.x] += red[threadIdx.x + o];
    __syncthreads();
  }
  if (threadIdx.x == 0) Z[n * 10 + k] = red[0] + bfc[k];
}

// Output LIF over [T,B,10]
__global__ void lif_out(const float* __restrict__ Z, float* __restrict__ out) {
  int tid = threadIdx.x;           // b*10+k
  if (tid >= BB * 10) return;
  float v = 0.f;
  for (int t = 0; t < TT; ++t) {
    float x = Z[t * (BB * 10) + tid];
    v = 0.5f * (v + x);
    float s = v >= 1.f ? 1.f : 0.f;
    out[t * (BB * 10) + tid] = s;
    v *= (1.f - s);
  }
}

// ---------------------------------------------------------------------------
extern "C" void kernel_launch(void* const* d_in, const int* in_sizes, int n_in,
                              void* d_out, int out_size, void* d_ws, size_t ws_size,
                              hipStream_t stream) {
  (void)in_sizes; (void)n_in; (void)out_size; (void)ws_size;
  const float* x   = (const float*)d_in[0];
  const float* w1  = (const float*)d_in[1];
  const float* wfc = (const float*)d_in[21];
  const float* bfc = (const float*)d_in[22];

  char* ws = (char*)d_ws;
  size_t off = 0;
  auto alloc = [&](size_t bytes) {
    size_t o = off; off += (bytes + 255) & ~(size_t)255; return o;
  };
  _Float16* Wp1 = (_Float16*)(ws + alloc(128L *  9 *  32 * 2));
  _Float16* Wp2 = (_Float16*)(ws + alloc(128L *  9 * 128 * 2));
  _Float16* Wp3 = (_Float16*)(ws + alloc(128L *  9 * 128 * 2));
  _Float16* Wp4 = (_Float16*)(ws + alloc(128L *  9 * 128 * 2));
  float* bnc1 = (float*)(ws + alloc(256 * 4));
  float* bnc2 = (float*)(ws + alloc(256 * 4));
  float* bnc3 = (float*)(ws + alloc(256 * 4));
  float* bnc4 = (float*)(ws + alloc(256 * 4));
  _Float16* X0 = (_Float16*)(ws + alloc(64L * 128 * 128 *  32 * 2));
  _Float16* X1 = (_Float16*)(ws + alloc(64L *  64 *  64 * 128 * 2));
  _Float16* X2 = (_Float16*)(ws + alloc(64L *  32 *  32 * 128 * 2));
  _Float16* X3 = (_Float16*)(ws + alloc(64L *  16 *  16 * 128 * 2));
  _Float16* X4 = (_Float16*)(ws + alloc(64L *   8 *   8 * 128 * 2));
  float* Z     = (float*)   (ws + alloc(64L * 10 * 4));
  float* U     = (float*)   (ws + alloc(64L * 128 * 128 * 128 * 4)); // reused all blocks

  // Prep: packed input, packed weights, BN constants
  pack_input<<<131072, 256, 0, stream>>>(x, X0);
  pack_w<<<144, 256, 0, stream>>>(w1, Wp1, 2, 32);
  pack_w<<<576, 256, 0, stream>>>((const float*)d_in[6],  Wp2, 128, 128);
  pack_w<<<576, 256, 0, stream>>>((const float*)d_in[11], Wp3, 128, 128);
  pack_w<<<576, 256, 0, stream>>>((const float*)d_in[16], Wp4, 128, 128);
  bn_prep<<<1, 128, 0, stream>>>((const float*)d_in[2],  (const float*)d_in[3],
                                 (const float*)d_in[4],  (const float*)d_in[5],  bnc1);
  bn_prep<<<1, 128, 0, stream>>>((const float*)d_in[7],  (const float*)d_in[8],
                                 (const float*)d_in[9],  (const float*)d_in[10], bnc2);
  bn_prep<<<1, 128, 0, stream>>>((const float*)d_in[12], (const float*)d_in[13],
                                 (const float*)d_in[14], (const float*)d_in[15], bnc3);
  bn_prep<<<1, 128, 0, stream>>>((const float*)d_in[17], (const float*)d_in[18],
                                 (const float*)d_in[19], (const float*)d_in[20], bnc4);

  // Block 1: conv(2->128 padded to 32) @128x128 (strips of 64 px), -> 64x64
  conv_wmma<32, 128, 4><<<64 * 128 * (128 / 64), 256, 0, stream>>>(X0, Wp1, bnc1, U);
  lif_pool<128><<<4L * 64 * 64 * 128 / 256, 256, 0, stream>>>(U, X1);
  // Block 2: conv(128->128) @64x64 -> 32x32
  conv_wmma<128, 64, 4><<<64 * 64 * (64 / 64), 256, 0, stream>>>(X1, Wp2, bnc2, U);
  lif_pool<64><<<4L * 32 * 32 * 128 / 256, 256, 0, stream>>>(U, X2);
  // Block 3: strips of 32 px
  conv_wmma<128, 32, 2><<<64 * 32 * (32 / 32), 256, 0, stream>>>(X2, Wp3, bnc3, U);
  lif_pool<32><<<4L * 16 * 16 * 128 / 256, 256, 0, stream>>>(U, X3);
  // Block 4: strips of 16 px
  conv_wmma<128, 16, 1><<<64 * 16 * (16 / 16), 256, 0, stream>>>(X3, Wp4, bnc4, U);
  lif_pool<16><<<4L * 8 * 8 * 128 / 256, 256, 0, stream>>>(U, X4);

  // FC + output LIF
  fc_kernel<<<640, 256, 0, stream>>>(X4, wfc, bfc, Z);
  lif_out<<<1, 64, 0, stream>>>(Z, (float*)d_out);
}
